// EGNNLayerV40_17068200034776
// MI455X (gfx1250) — matrix operand order repaired and checked by reading, not compiled, not run on hardware
//
#include <hip/hip_runtime.h>

// ---------------------------------------------------------------------------
// CDNA5 (gfx1250) fused EGNN layer: f16 WMMA (16x16x32, f32 accum)
// 64-edge tiles, register-level B-fragment reuse across 4 row-blocks.
// ---------------------------------------------------------------------------

typedef __attribute__((ext_vector_type(16))) _Float16 v16h;
typedef __attribute__((ext_vector_type(8)))  float    v8f;

#define TILE_E   64      // edges per workgroup (4 row-blocks of 16)
#define NRB      4       // TILE_E / 16
#define NODE_D   128
#define HID_D    256
#define EDGE_D   64
#define M_IN     320     // 2*NODE_D + EDGE_D
#define NKB_M    10      // 320/32 K-blocks of m_input
#define NKB_H    8       // 256/32 K-blocks of hid
#define NKB_EV   2       // 64/32  K-blocks of evec

// Workspace layout in _Float16 units (B-fragment tiles: 512 f16 = 32 lanes x 16)
#define WS_WN1   0
#define WS_WC1   (WS_WN1 + NKB_M*16*512)   // Wn1: 10 kb x 16 cb tiles
#define WS_WN2   (WS_WC1 + NKB_M*16*512)   // Wc1: 10 kb x 16 cb tiles
#define WS_WE2   (WS_WN2 + NKB_H*8*512)    // Wn2: 8 kb x 8 cb tiles
// We2: 2 kb x 4 cb tiles

__device__ __forceinline__ float silu_f(float v) {
    return v / (1.0f + __expf(-v));
}

// ISA 16-bit A-matrix fragment mapping (16x32 tile):
// lanes 0-15 hold K in {0..7, 16..23}; lanes 16-31 hold K in {8..15, 24..31}.
__device__ __forceinline__ int afrag_off(int kb, int row, int kk) {
    int hf  = (kk >> 3) & 1;
    int kkp = kk - (hf << 3);
    int e   = (kkp < 8) ? kkp : (kkp - 8);
    return kb * 512 + ((hf << 4) + row) * 16 + e;
}

__device__ __forceinline__ v8f wmma16(v16h a, v16h b, v8f c) {
    return __builtin_amdgcn_wmma_f32_16x16x32_f16(false, a, false, b,
                                                  (short)0, c, false, false);
}

// ---------------------------------------------------------------------------
// Pack a row-major f32 weight [K x N] into f16 B-fragment tiles.
// B layout: lane L holds column n = L%16; elem e -> K = 8*(L/16) + e (e<8), 16 + 8*(L/16) + (e-8) (e>=8).
// ---------------------------------------------------------------------------
__global__ void pack_b_kernel(const float* __restrict__ W,
                              _Float16* __restrict__ out, int K, int N) {
    int ncb   = N >> 4;
    int total = (K >> 5) * ncb * 512;
    for (int tid = blockIdx.x * blockDim.x + threadIdx.x; tid < total;
         tid += gridDim.x * blockDim.x) {
        int tile = tid >> 9, r = tid & 511;
        int lane = r >> 4, e = r & 15;
        int kb = tile / ncb, cb = tile - kb * ncb;
        int k = kb * 32 + ((lane >> 4) << 3) + (e & 7) + ((e >> 3) << 4);
        int n = cb * 16 + (lane & 15);
        out[tid] = (_Float16)W[k * N + n];
    }
}

// d_out = [h | x] (residual base for atomic accumulation)
__global__ void init_out_kernel(const float* __restrict__ h,
                                const float* __restrict__ x,
                                float* __restrict__ out, int nh, int nx) {
    for (int i = blockIdx.x * blockDim.x + threadIdx.x; i < nh + nx;
         i += gridDim.x * blockDim.x)
        out[i] = (i < nh) ? h[i] : x[i - nh];
}

// ---------------------------------------------------------------------------
// Main fused edge kernel: one 64-edge tile per workgroup, 8 wave32s.
// ---------------------------------------------------------------------------
__global__ __launch_bounds__(256)
void egnn_edge_kernel(const float* __restrict__ h, const float* __restrict__ x,
                      const int* __restrict__ ei, const float* __restrict__ edist,
                      int E,
                      const float* __restrict__ bn1, const float* __restrict__ bn2,
                      const float* __restrict__ bc1, const float* __restrict__ Wc2,
                      const float* __restrict__ We1, const float* __restrict__ be1,
                      const float* __restrict__ be2,
                      const _Float16* __restrict__ ws,
                      float* __restrict__ hout, float* __restrict__ xout) {
    // m_input fragments: 64 rows x 320 K  (40 KB)
    __shared__ alignas(32) _Float16 Am[NRB * NKB_M * 512];
    // hid fragments: 64 rows x 256 K (32 KB); first 8 KB double as evec frags
    // (evec is consumed in GEMM0, before the barrier that precedes hid stores)
    __shared__ alignas(32) _Float16 Ah[NRB * NKB_H * 512];
    __shared__ int   s_src[TILE_E], s_dst[TILE_E];
    __shared__ float s_cw[TILE_E];

    const int t    = threadIdx.x;
    const int wave = t >> 5;
    const int lane = t & 31;
    const int hf   = lane >> 4;     // C/D half: rows i or i+8
    const int nloc = lane & 15;     // C/D column within tile
    const int tileBase = (int)blockIdx.x * TILE_E;

    // ---- 1. edge meta -----------------------------------------------------
    if (t < TILE_E) {
        int eg = tileBase + t;
        int s = 0, d = 0;
        if (eg < E) { s = ei[eg]; d = ei[E + eg]; }
        s_src[t] = s; s_dst[t] = d; s_cw[t] = 0.0f;
    }
    __syncthreads();

    // ---- 2. gather h[src], h[dst] -> Am (k = 0..255) ----------------------
    for (int idx = t; idx < TILE_E * 64; idx += 256) {
        int edge = idx >> 6, k4 = idx & 63;
        int row = edge & 15, rb = edge >> 4;
        int node  = (k4 < 32) ? s_src[edge] : s_dst[edge];
        int kbase = (k4 & 31) << 2;
        const float4 v = *(const float4*)&h[node * NODE_D + kbase];
        int km0 = (k4 < 32) ? kbase : (NODE_D + kbase);
        float vv[4] = {v.x, v.y, v.z, v.w};
#pragma unroll
        for (int j = 0; j < 4; ++j) {
            int km = km0 + j;
            Am[rb * (NKB_M * 512) + afrag_off(km >> 5, row, km & 31)] = (_Float16)vv[j];
        }
    }

    // ---- 3. evec = silu(dist*We1 + be1) -> Ah[0..4095] (aliased) ----------
    for (int idx = t; idx < TILE_E * EDGE_D; idx += 256) {
        int edge = idx >> 6, k = idx & 63;
        int row = edge & 15, rb = edge >> 4;
        int eg = tileBase + edge;
        float dv  = (eg < E) ? edist[eg] : 0.0f;
        float val = silu_f(dv * We1[k] + be1[k]);
        Ah[rb * (NKB_EV * 512) + afrag_off(k >> 5, row, k & 31)] = (_Float16)val;
    }
    __syncthreads();

    // ---- 4. GEMM0: edge_attr = evec @ We2 + be2 -> Am (k = 256..319) ------
    {
        int cb = wave & 3;
        int rbase = (wave >> 2) * 2;      // rows {0,1} or {2,3}
        v8f c[2] = {};
#pragma unroll
        for (int kb = 0; kb < NKB_EV; ++kb) {
            v16h b = *(const v16h*)&ws[WS_WE2 + (kb * 4 + cb) * 512 + lane * 16];
#pragma unroll
            for (int r = 0; r < 2; ++r) {
                v16h a = *(const v16h*)&Ah[(rbase + r) * (NKB_EV * 512) + kb * 512 + lane * 16];
                c[r] = wmma16(a, b, c[r]);
            }
        }
        int col = cb * 16 + nloc;
        float bias = be2[col];
        int km = 2 * NODE_D + col;
#pragma unroll
        for (int r = 0; r < 2; ++r) {
#pragma unroll
            for (int i = 0; i < 8; ++i) {
                int row = i + (hf << 3);
                Am[(rbase + r) * (NKB_M * 512) + afrag_off(km >> 5, row, km & 31)] =
                    (_Float16)(c[r][i] + bias);
            }
        }
    }
    __syncthreads();   // Am complete; evec region dead -> Ah reusable

    // ---- 5. fused GEMM1+GEMM3 over all 4 row-blocks per B fragment --------
    // hid = silu(m@Wn1+bn1) -> Ah ; cw += silu(m@Wc1+bc1)*Wc2
    for (int cbi = 0; cbi < 2; ++cbi) {
        int cb = wave * 2 + cbi;          // 16 cbs / 8 waves
        v8f cn[NRB] = {};
        v8f cc[NRB] = {};
#pragma unroll
        for (int kb = 0; kb < NKB_M; ++kb) {
            v16h bN = *(const v16h*)&ws[WS_WN1 + (kb * 16 + cb) * 512 + lane * 16];
            v16h bC = *(const v16h*)&ws[WS_WC1 + (kb * 16 + cb) * 512 + lane * 16];
            if (kb + 1 < NKB_M) {
                __builtin_prefetch(&ws[WS_WN1 + ((kb + 1) * 16 + cb) * 512 + lane * 16], 0, 1);
                __builtin_prefetch(&ws[WS_WC1 + ((kb + 1) * 16 + cb) * 512 + lane * 16], 0, 1);
            }
#pragma unroll
            for (int rb = 0; rb < NRB; ++rb) {
                v16h a = *(const v16h*)&Am[rb * (NKB_M * 512) + kb * 512 + lane * 16];
                cn[rb] = wmma16(a, bN, cn[rb]);
                cc[rb] = wmma16(a, bC, cc[rb]);
            }
        }
        int col = cb * 16 + nloc;
        float b1 = bn1[col], b2 = bc1[col], w2 = Wc2[col];
#pragma unroll
        for (int rb = 0; rb < NRB; ++rb) {
#pragma unroll
            for (int i = 0; i < 8; ++i) {
                int row = i + (hf << 3);
                float hv = silu_f(cn[rb][i] + b1);
                Ah[rb * (NKB_H * 512) + afrag_off(col >> 5, row, col & 31)] = (_Float16)hv;
                atomicAdd(&s_cw[rb * 16 + row], silu_f(cc[rb][i] + b2) * w2);
            }
        }
    }
    __syncthreads();

    // ---- 6. GEMM2: m = hid @ Wn2 + bn2, scatter-add into h_out ------------
    {
        int cb = wave;                    // 8 cbs / 8 waves
        v8f c[NRB] = {};
#pragma unroll
        for (int kb = 0; kb < NKB_H; ++kb) {
            v16h b = *(const v16h*)&ws[WS_WN2 + (kb * 8 + cb) * 512 + lane * 16];
#pragma unroll
            for (int rb = 0; rb < NRB; ++rb) {
                v16h a = *(const v16h*)&Ah[rb * (NKB_H * 512) + kb * 512 + lane * 16];
                c[rb] = wmma16(a, b, c[rb]);
            }
        }
        int col = cb * 16 + nloc;
        float bias = bn2[col];
#pragma unroll
        for (int rb = 0; rb < NRB; ++rb) {
#pragma unroll
            for (int i = 0; i < 8; ++i) {
                int r  = rb * 16 + i + (hf << 3);
                int eg = tileBase + r;
                if (eg < E)
                    atomicAdd(&hout[s_dst[r] * NODE_D + col], c[rb][i] + bias);
            }
        }
    }

    // ---- 7. coordinate update (s_cw finalized at previous barrier) --------
    if (t < TILE_E) {
        int eg = tileBase + t;
        if (eg < E) {
            float wgt = s_cw[t];
            int s = s_src[t], d = s_dst[t];
#pragma unroll
            for (int j = 0; j < 3; ++j) {
                float dv = x[s * 3 + j] - x[d * 3 + j];
                atomicAdd(&xout[d * 3 + j], wgt * dv);
            }
        }
    }
}

// ---------------------------------------------------------------------------
// Launch
// ---------------------------------------------------------------------------
extern "C" void kernel_launch(void* const* d_in, const int* in_sizes, int n_in,
                              void* d_out, int out_size, void* d_ws, size_t ws_size,
                              hipStream_t stream) {
    const float* h     = (const float*)d_in[0];
    const float* x     = (const float*)d_in[1];
    const int*   ei    = (const int*)d_in[2];
    const float* edist = (const float*)d_in[3];
    const float* Wn1   = (const float*)d_in[4];
    const float* bn1   = (const float*)d_in[5];
    const float* Wn2   = (const float*)d_in[6];
    const float* bn2   = (const float*)d_in[7];
    const float* Wc1   = (const float*)d_in[8];
    const float* bc1   = (const float*)d_in[9];
    const float* Wc2   = (const float*)d_in[10];
    const float* We1   = (const float*)d_in[11];
    const float* be1   = (const float*)d_in[12];
    const float* We2   = (const float*)d_in[13];
    const float* be2   = (const float*)d_in[14];

    int E  = in_sizes[3];        // N_EDGES
    int nh = in_sizes[0];        // N_NODES * NODE_DIM
    int nx = in_sizes[1];        // N_NODES * 3

    float*    out = (float*)d_out;
    _Float16* ws  = (_Float16*)d_ws;

    // Pack weights into f16 B-fragment tiles (L2-resident across main kernel).
    pack_b_kernel<<<64, 256, 0, stream>>>(Wn1, ws + WS_WN1, M_IN,  HID_D);
    pack_b_kernel<<<64, 256, 0, stream>>>(Wc1, ws + WS_WC1, M_IN,  HID_D);
    pack_b_kernel<<<32, 256, 0, stream>>>(Wn2, ws + WS_WN2, HID_D, NODE_D);
    pack_b_kernel<<< 8, 256, 0, stream>>>(We2, ws + WS_WE2, EDGE_D, EDGE_D);

    // Residual base: d_out = [h | x]; edge kernel accumulates atomically.
    init_out_kernel<<<1024, 256, 0, stream>>>(h, x, out, nh, nx);

    int ntiles = (E + TILE_E - 1) / TILE_E;
    egnn_edge_kernel<<<ntiles, 256, 0, stream>>>(
        h, x, ei, edist, E, bn1, bn2, bc1, Wc2, We1, be1, be2,
        ws, out, out + nh);
}
